// SEALNet_35175782154595
// MI455X (gfx1250) — compile-verified, matrix operand order checked by reading, not run on hardware
//
#include <hip/hip_runtime.h>
#include <hip/hip_bf16.h>
#include <math.h>
#include <stdint.h>

// ---------------------------------------------------------------------------
// SEAL / DGCNN inference pipeline for gfx1250 (MI455X).
// GEMMs: TDM (tensor_load_to_lds) stages A tiles into padded LDS,
//        v_wmma_f32_16x16x32_bf16 does the math (bf16 in, f32 accumulate).
// ---------------------------------------------------------------------------

typedef __attribute__((ext_vector_type(16))) __bf16 v16bf;
typedef __attribute__((ext_vector_type(8)))  float  v8f;
typedef unsigned int u32x4 __attribute__((ext_vector_type(4)));
typedef int          i32x8 __attribute__((ext_vector_type(8)));
typedef int          i32x4 __attribute__((ext_vector_type(4)));

#define XD 97   // concat latent dim
#define KP 60   // sort-pool k
#define HC 32   // hidden channels

// ---------------- utility fills ----------------
__global__ void fill_f32(float* p, float v, long n) {
  long i = (long)blockIdx.x * blockDim.x + threadIdx.x;
  if (i < n) p[i] = v;
}
__global__ void fill_i32(int* p, int v, long n) {
  long i = (long)blockIdx.x * blockDim.x + threadIdx.x;
  if (i < n) p[i] = v;
}

// ---------------- degree / norm ----------------
__global__ void deg_accum(const int* __restrict__ ei, long E, float* deg) {
  long e = (long)blockIdx.x * blockDim.x + threadIdx.x;
  if (e >= E) return;
  int dst = ei[E + e];
  atomicAdd(&deg[dst], 1.0f);
}
__global__ void rsqrt_inplace(float* p, long n) {
  long i = (long)blockIdx.x * blockDim.x + threadIdx.x;
  if (i < n) p[i] = rsqrtf(p[i]);
}

// ---------------- bf16 WMMA GEMM with TDM A-tile staging -------------------
// C[M x Nc] = A[M x K] * B[K x Nc].  K compile-time (full unroll).
// Block: 256 threads = 8 waves; wave -> (rowTile 0..3, colTile16 0..1).
// Grid:  x = ceil(M/64), y = Nc/32.
// Wave 0 issues one TDM descriptor per K-chunk: loads 64 x CHUNK f32 tile of A
// into LDS with 4-DWORD padding every CHUNK DWORDs (bank-conflict-free rows,
// zero-filled out-of-range rows via tensor_dim1 = rows remaining).
template <int K>
__global__ __launch_bounds__(256)
void gemm_bf16_wmma(const float* __restrict__ A, int lda, int M,
                    const float* __restrict__ B, int ldb,
                    float* __restrict__ C, int ldc)
{
  constexpr int CHUNK = (K % 128 == 0) ? 128 : ((K % 64 == 0) ? 64 : 32);
  constexpr int LROW  = CHUNK + 4;                       // padded LDS row (f32)
  constexpr unsigned PADIV = (CHUNK == 128) ? 6u : ((CHUNK == 64) ? 5u : 4u);
  __shared__ float aTile[64 * LROW];

  const int wave = threadIdx.x >> 5;
  const int lane = threadIdx.x & 31;
  const int rowBase = blockIdx.x * 64 + (wave >> 1) * 16;
  const int colBase = blockIdx.y * 32 + (wave & 1) * 16;
  const int ncol = colBase + (lane & 15);
  const int lo   = (lane < 16) ? 0 : 8;                  // K-half per ISA layout
  const int lrow = (wave >> 1) * 16 + (lane & 15);       // local A row 0..63

  const unsigned ldsBase = (unsigned)(uintptr_t)&aTile[0];
  const long blockRow = (long)blockIdx.x * 64;
  long rowsRem = (long)M - blockRow;
  if (rowsRem < 0) rowsRem = 0;

  v8f acc = {};
#pragma unroll
  for (int kc = 0; kc < K; kc += CHUNK) {
    // ---- TDM: DMA A[blockRow .. +64, kc .. kc+CHUNK) -> padded LDS tile ----
    if (threadIdx.x < 32) {                               // wave-uniform branch
      unsigned long long ga =
          (unsigned long long)(uintptr_t)(A + blockRow * (long)lda + kc);
      u32x4 g0 = { 1u,                                    // count = 1
                   ldsBase,                               // lds_addr
                   (unsigned)ga,                          // global_addr[31:0]
                   (unsigned)((ga >> 32) & 0x1FFFFFFull) | 0x80000000u }; // type=2
      unsigned rr = (unsigned)rowsRem;
      unsigned uld = (unsigned)lda;
      i32x8 g1 = {
        (int)((2u << 16) | (1u << 20) | (PADIV << 22) | (3u << 25)), // data_size=4B, pad on
        (int)((uld & 0xFFFFu) << 16),                    // tensor_dim0 lo
        (int)(((uld >> 16) & 0xFFFFu) | ((rr & 0xFFFFu) << 16)),     // dim0 hi | dim1 lo
        (int)(((rr >> 16) & 0xFFFFu) | ((unsigned)CHUNK << 16)),     // dim1 hi | tile_dim0
        (int)64u,                                        // tile_dim1 = 64 rows
        (int)uld,                                        // dim0_stride lo
        0, 0 };
      i32x4 z4 = { 0, 0, 0, 0 };
      i32x8 z8 = { 0, 0, 0, 0, 0, 0, 0, 0 };
      __builtin_amdgcn_tensor_load_to_lds(g0, g1, z4, z4, z8, 0);
      __builtin_amdgcn_s_wait_tensorcnt(0);
    }
    __syncthreads();

#pragma unroll
    for (int k0 = 0; k0 < CHUNK; k0 += 32) {
      // A fragment from LDS: lane&15 = row; elems 0..7 -> k0+lo+i,
      // elems 8..15 -> k0+16+lo+i  (16x32 bf16 layout).
      const float* ar = &aTile[lrow * LROW + k0];
      v16bf a;
#pragma unroll
      for (int i = 0; i < 8; ++i) {
        a[i]     = (__bf16)ar[lo + i];
        a[8 + i] = (__bf16)ar[16 + lo + i];
      }
      // B fragment from global (tiny, cache-resident): lane&15 = col.
      v16bf b;
#pragma unroll
      for (int i = 0; i < 8; ++i) {
        b[i]     = (__bf16)B[(long)(kc + k0 + lo + i)      * ldb + ncol];
        b[8 + i] = (__bf16)B[(long)(kc + k0 + 16 + lo + i) * ldb + ncol];
      }
      acc = __builtin_amdgcn_wmma_f32_16x16x32_bf16(
          false, a, false, b, (short)0, acc, false, false);
    }
    __syncthreads();   // tile reuse barrier before next TDM overwrite
  }

  // C/D layout: lane<16 -> rows 0..7, lane>=16 -> rows 8..15; col = lane&15.
  const int rbase = (lane < 16) ? 0 : 8;
#pragma unroll
  for (int r = 0; r < 8; ++r) {
    int m = rowBase + rbase + r;
    if (m < M) C[(long)m * ldc + ncol] = acc[r];
  }
}

// ---------------- edge propagation (H channels, 4 at a time) ---------------
__global__ void edge_prop4(const int* __restrict__ ei, long E,
                           const float* __restrict__ dis,
                           const float* __restrict__ t, int H,
                           float* __restrict__ agg)
{
  const int groups = H >> 2;
  long idx = (long)blockIdx.x * blockDim.x + threadIdx.x;
  long total = E * groups;
  if (idx >= total) return;
  long e = idx / groups;
  int  c = (int)(idx - e * groups) * 4;
  int src = ei[e];
  int dst = ei[E + e];
  float norm = dis[src] * dis[dst];
  const float4* tp = (const float4*)(t + (long)src * H + c);
  __builtin_prefetch(tp, 0, 1);          // global_prefetch on the gather stream
  float4 v = *tp;
  float* ap = agg + (long)dst * H + c;
  atomicAdd(ap + 0, v.x * norm);
  atomicAdd(ap + 1, v.y * norm);
  atomicAdd(ap + 2, v.z * norm);
  atomicAdd(ap + 3, v.w * norm);
}

__global__ void edge_prop1(const int* __restrict__ ei, long E,
                           const float* __restrict__ dis,
                           const float* __restrict__ t1,
                           float* __restrict__ agg)
{
  long e = (long)blockIdx.x * blockDim.x + threadIdx.x;
  if (e >= E) return;
  int src = ei[e];
  int dst = ei[E + e];
  atomicAdd(&agg[dst], t1[src] * dis[src] * dis[dst]);
}

// -------- self-loop + bias + tanh, write into xcat[:, colOff .. colOff+H) --
__global__ void self_act(const float* __restrict__ t,
                         const float* __restrict__ agg,
                         const float* __restrict__ dis,
                         const float* __restrict__ bias, int H, long N,
                         float* __restrict__ xcat, int colOff)
{
  long idx = (long)blockIdx.x * blockDim.x + threadIdx.x;
  if (idx >= N * H) return;
  long n = idx / H;
  int  c = (int)(idx - n * H);
  float d = dis[n];
  float v = agg[idx] + t[idx] * d * d + bias[c];
  xcat[n * XD + colOff + c] = tanhf(v);
}

// ---------------- layer 3: 32 -> 1 projection ------------------------------
__global__ void proj1(const float* __restrict__ xcat,
                      const float* __restrict__ W3,
                      float* __restrict__ t1, long N)
{
  long n = (long)blockIdx.x * blockDim.x + threadIdx.x;
  if (n >= N) return;
  const float* r = xcat + n * XD + 64;
  float s = 0.f;
#pragma unroll
  for (int c = 0; c < HC; ++c) s += r[c] * W3[c];
  t1[n] = s;
}

// ---------------- per-graph offsets ----------------------------------------
__global__ void batch_count(const int* __restrict__ batch, long N, int* counts) {
  long i = (long)blockIdx.x * blockDim.x + threadIdx.x;
  if (i < N) atomicAdd(&counts[batch[i]], 1);
}
__global__ void scan_offsets(const int* __restrict__ counts, int* offs, int G) {
  if (blockIdx.x == 0 && threadIdx.x == 0) {
    int acc = 0;
    offs[0] = 0;
    for (int g = 0; g < G; ++g) { acc += counts[g]; offs[g + 1] = acc; }
  }
}

// ---------------- sort-pool: per-graph top-60 by last channel --------------
#define MAXS 512
__global__ __launch_bounds__(256)
void sort_pool(const float* __restrict__ xcat, const int* __restrict__ offs,
               int* __restrict__ topidx)
{
  __shared__ float sval[MAXS];
  __shared__ int   sidx[MAXS];
  int g = blockIdx.x;
  int start = offs[g], end = offs[g + 1];
  int cnt = end - start; if (cnt > MAXS) cnt = MAXS;
  for (int i = threadIdx.x; i < MAXS; i += blockDim.x) {
    if (i < cnt) {
      int node = start + i;
      sval[i] = xcat[(long)node * XD + (XD - 1)];
      sidx[i] = node;
    } else { sval[i] = -3.0e38f; sidx[i] = -1; }
  }
  __syncthreads();
  // bitonic sort, descending
  for (int ksz = 2; ksz <= MAXS; ksz <<= 1) {
    for (int j = ksz >> 1; j > 0; j >>= 1) {
      for (int i = threadIdx.x; i < MAXS; i += blockDim.x) {
        int ixj = i ^ j;
        if (ixj > i) {
          bool desc = ((i & ksz) == 0);
          bool sw = desc ? (sval[i] < sval[ixj]) : (sval[i] > sval[ixj]);
          if (sw) {
            float tv = sval[i]; sval[i] = sval[ixj]; sval[ixj] = tv;
            int   ti = sidx[i]; sidx[i] = sidx[ixj]; sidx[ixj] = ti;
          }
        }
      }
      __syncthreads();
    }
  }
  for (int k = threadIdx.x; k < KP; k += blockDim.x)
    topidx[(long)g * KP + k] = (k < cnt) ? sidx[k] : -1;
}

// ---------------- per-graph CNN head: conv1+pool+conv2 -> f[G,832] ---------
__global__ __launch_bounds__(128)
void graph_feats(const float* __restrict__ xcat, const int* __restrict__ topidx,
                 const float* __restrict__ cw1, const float* __restrict__ cb1,
                 const float* __restrict__ cw2, const float* __restrict__ cb2,
                 float* __restrict__ f)
{
  __shared__ float z[KP][XD + 1];   // 60 x 98 (padded)
  __shared__ float c1[16][KP];      // 16 x 60
  __shared__ float pbuf[16][30];
  int g = blockIdx.x;

  for (int t = threadIdx.x; t < KP * XD; t += blockDim.x) {
    int k = t / XD, d = t - k * XD;
    int node = topidx[(long)g * KP + k];
    z[k][d] = (node >= 0) ? xcat[(long)node * XD + d] : 0.f;
  }
  __syncthreads();

  for (int t = threadIdx.x; t < 16 * KP; t += blockDim.x) {
    int o = t / KP, k = t - o * KP;
    float s = cb1[o];
    for (int d = 0; d < XD; ++d) s += z[k][d] * cw1[o * XD + d];
    c1[o][k] = fmaxf(s, 0.f);
  }
  __syncthreads();

  for (int t = threadIdx.x; t < 16 * 30; t += blockDim.x) {
    int o = t / 30, j = t - o * 30;
    pbuf[o][j] = fmaxf(c1[o][2 * j], c1[o][2 * j + 1]);
  }
  __syncthreads();

  for (int t = threadIdx.x; t < 32 * 26; t += blockDim.x) {
    int o = t / 26, tt = t - o * 26;
    float s = cb2[o];
#pragma unroll
    for (int i = 0; i < 16; ++i)
#pragma unroll
      for (int kw = 0; kw < 5; ++kw)
        s += pbuf[i][tt + kw] * cw2[(o * 16 + i) * 5 + kw];
    f[(long)g * 832 + o * 26 + tt] = fmaxf(s, 0.f);
  }
}

// ---------------- bias + relu on dense output ------------------------------
__global__ void relu_bias(float* __restrict__ Cm, const float* __restrict__ bias,
                          long total, int ldc)
{
  long i = (long)blockIdx.x * blockDim.x + threadIdx.x;
  if (i >= total) return;
  int c = (int)(i % ldc);
  Cm[i] = fmaxf(Cm[i] + bias[c], 0.f);
}

// ---------------- 128 -> 4 + softmax ---------------------------------------
__global__ void head(const float* __restrict__ hd, const float* __restrict__ lw2,
                     const float* __restrict__ lb2, float* __restrict__ out, int G)
{
  int g = blockIdx.x * blockDim.x + threadIdx.x;
  if (g >= G) return;
  const float* hr = hd + (long)g * 128;
  float logit[4];
#pragma unroll
  for (int c = 0; c < 4; ++c) {
    float s = lb2[c];
    for (int i = 0; i < 128; ++i) s += hr[i] * lw2[i * 4 + c];
    logit[c] = s;
  }
  float mx = logit[0];
#pragma unroll
  for (int c = 1; c < 4; ++c) mx = fmaxf(mx, logit[c]);
  float sum = 0.f, e[4];
#pragma unroll
  for (int c = 0; c < 4; ++c) { e[c] = expf(logit[c] - mx); sum += e[c]; }
  float inv = 1.0f / sum;
#pragma unroll
  for (int c = 0; c < 4; ++c) out[(long)g * 4 + c] = e[c] * inv;
}

// ===========================================================================
extern "C" void kernel_launch(void* const* d_in, const int* in_sizes, int n_in,
                              void* d_out, int out_size, void* d_ws, size_t ws_size,
                              hipStream_t stream)
{
  const float* x     = (const float*)d_in[0];
  const int*   ei    = (const int*)  d_in[1];
  const int*   batch = (const int*)  d_in[2];
  const float* W0 = (const float*)d_in[3];
  const float* b0 = (const float*)d_in[4];
  const float* W1 = (const float*)d_in[5];
  const float* b1 = (const float*)d_in[6];
  const float* W2 = (const float*)d_in[7];
  const float* b2 = (const float*)d_in[8];
  const float* W3 = (const float*)d_in[9];
  const float* b3 = (const float*)d_in[10];
  const float* cw1 = (const float*)d_in[11];
  const float* cb1 = (const float*)d_in[12];
  const float* cw2 = (const float*)d_in[13];
  const float* cb2 = (const float*)d_in[14];
  const float* lw1 = (const float*)d_in[15];
  const float* lb1 = (const float*)d_in[16];
  const float* lw2 = (const float*)d_in[17];
  const float* lb2 = (const float*)d_in[18];

  const long N = in_sizes[2];
  const int  F = (int)(in_sizes[0] / N);     // 128
  const long E = in_sizes[1] / 2;            // 8M
  const int  G = out_size / 4;               // 5000
  const int  H = HC;

  // ---- carve workspace ----
  char* wp = (char*)d_ws;
  auto carve = [&](size_t bytes) -> void* {
    void* r = (void*)wp;
    wp += (bytes + 255) & ~(size_t)255;
    return r;
  };
  float* dis  = (float*)carve(sizeof(float) * N);
  float* t32  = (float*)carve(sizeof(float) * N * H);
  float* agg  = (float*)carve(sizeof(float) * N * H);
  float* xcat = (float*)carve(sizeof(float) * N * XD);
  float* t1   = (float*)carve(sizeof(float) * N);
  int* counts = (int*)carve(sizeof(int) * G);
  int* offs   = (int*)carve(sizeof(int) * (G + 1));
  int* topidx = (int*)carve(sizeof(int) * (size_t)G * KP);
  float* fbuf = (float*)carve(sizeof(float) * (size_t)G * 832);
  float* hdn  = (float*)carve(sizeof(float) * (size_t)G * 128);

  auto blks = [](long n, int bs) { return (unsigned)((n + bs - 1) / bs); };

  // ---- degrees / norm ----
  fill_f32<<<blks(N, 256), 256, 0, stream>>>(dis, 1.0f, N);
  deg_accum<<<blks(E, 256), 256, 0, stream>>>(ei, E, dis);
  rsqrt_inplace<<<blks(N, 256), 256, 0, stream>>>(dis, N);

  const long NH = N * H;
  const unsigned gemmRows = blks(N, 64);

  // ---- layer 0: x[N,128] @ W0[128,32] ----
  gemm_bf16_wmma<128><<<dim3(gemmRows, 1), 256, 0, stream>>>(x, F, (int)N, W0, H, t32, H);
  fill_f32<<<blks(NH, 256), 256, 0, stream>>>(agg, 0.f, NH);
  edge_prop4<<<blks(E * (H / 4), 256), 256, 0, stream>>>(ei, E, dis, t32, H, agg);
  self_act<<<blks(NH, 256), 256, 0, stream>>>(t32, agg, dis, b0, H, N, xcat, 0);

  // ---- layer 1: xcat[:,0:32] @ W1 ----
  gemm_bf16_wmma<32><<<dim3(gemmRows, 1), 256, 0, stream>>>(xcat, XD, (int)N, W1, H, t32, H);
  fill_f32<<<blks(NH, 256), 256, 0, stream>>>(agg, 0.f, NH);
  edge_prop4<<<blks(E * (H / 4), 256), 256, 0, stream>>>(ei, E, dis, t32, H, agg);
  self_act<<<blks(NH, 256), 256, 0, stream>>>(t32, agg, dis, b1, H, N, xcat, 32);

  // ---- layer 2: xcat[:,32:64] @ W2 ----
  gemm_bf16_wmma<32><<<dim3(gemmRows, 1), 256, 0, stream>>>(xcat + 32, XD, (int)N, W2, H, t32, H);
  fill_f32<<<blks(NH, 256), 256, 0, stream>>>(agg, 0.f, NH);
  edge_prop4<<<blks(E * (H / 4), 256), 256, 0, stream>>>(ei, E, dis, t32, H, agg);
  self_act<<<blks(NH, 256), 256, 0, stream>>>(t32, agg, dis, b2, H, N, xcat, 64);

  // ---- layer 3: xcat[:,64:96] @ W3[32,1] ----
  proj1<<<blks(N, 256), 256, 0, stream>>>(xcat, W3, t1, N);
  fill_f32<<<blks(N, 256), 256, 0, stream>>>(agg, 0.f, N);
  edge_prop1<<<blks(E, 256), 256, 0, stream>>>(ei, E, dis, t1, agg);
  self_act<<<blks(N, 256), 256, 0, stream>>>(t1, agg, dis, b3, 1, N, xcat, 96);

  // ---- sort-pool ----
  fill_i32<<<blks(G, 256), 256, 0, stream>>>(counts, 0, G);
  batch_count<<<blks(N, 256), 256, 0, stream>>>(batch, N, counts);
  scan_offsets<<<1, 1, 0, stream>>>(counts, offs, G);
  sort_pool<<<G, 256, 0, stream>>>(xcat, offs, topidx);

  // ---- per-graph CNN ----
  graph_feats<<<G, 128, 0, stream>>>(xcat, topidx, cw1, cb1, cw2, cb2, fbuf);

  // ---- dense: f[G,832] @ lw1[832,128] (WMMA, 13 TDM-staged chunks) ----
  gemm_bf16_wmma<832><<<dim3(blks(G, 64), 4), 256, 0, stream>>>(fbuf, 832, G, lw1, 128, hdn, 128);
  relu_bias<<<blks((long)G * 128, 256), 256, 0, stream>>>(hdn, lb1, (long)G * 128, 128);

  // ---- classifier + softmax ----
  head<<<blks(G, 128), 128, 0, stream>>>(hdn, lw2, lb2, (float*)d_out, G);
}
